// MixedLlamaDecoderLayer_732
// MI455X (gfx1250) — matrix operand order repaired and checked by reading, not compiled, not run on hardware
//
// MixedLlamaDecoderLayer for MI455X (gfx1250, wave32, WMMA + TDM).
//
// Memory-bound analysis: fp32 weights total ~720MB -> ~31us @ 23.3TB/s HBM;
// 1.48 TFLOP of matmul is <1us at bf16 WMMA rates, so we stream fp32 weights
// exactly once (M-major grid + 192MB L2 keeps activations & weight panels hot),
// convert to bf16 during LDS staging, and accumulate in fp32 via
// v_wmma_f32_16x16x32_bf16. The GEMM is register-pipelined: next tile's global
// loads are issued before the WMMA block so HBM latency overlaps compute.
// Attention is flash-style with bf16 WMMA for Q.K^T and P.V, fp32 online
// softmax; the K tile is fetched by the Tensor Data Mover (tensor_load_to_lds,
// padded-row descriptor) overlapped with manual V^T staging (s_wait_tensorcnt).

#include <hip/hip_runtime.h>
#include <cstdint>
#include <cstddef>

#define B_    4
#define S_    1024
#define HID_  4096
#define NH_   32
#define NKV_  8
#define HD_   128
#define FF_   11008
#define TOK_  (B_ * S_)

typedef __bf16 bf16;
typedef __attribute__((ext_vector_type(16))) __bf16 v16bf;
typedef __attribute__((ext_vector_type(8)))  __bf16 v8bf;
typedef __attribute__((ext_vector_type(2)))  __bf16 v2bf;
typedef __attribute__((ext_vector_type(8)))  float  v8f;

// ---------------- fp32 -> bf16 (RNE) ----------------------------------------
#if __has_builtin(__builtin_amdgcn_cvt_pk_bf16_f32)
__device__ __forceinline__ v2bf f2bf2(float a, float b) {
  auto r = __builtin_amdgcn_cvt_pk_bf16_f32(a, b);   // v_cvt_pk_bf16_f32
  static_assert(sizeof(r) == sizeof(v2bf), "cvt_pk_bf16_f32 return size");
  return __builtin_bit_cast(v2bf, r);
}
#else
#warning "gfx1250 probe: __builtin_amdgcn_cvt_pk_bf16_f32 NOT available; using native __bf16 cast"
__device__ __forceinline__ v2bf f2bf2(float a, float b) {
  v2bf r; r[0] = (bf16)a; r[1] = (bf16)b;            // backend fptrunc lowering
  return r;
}
#endif
__device__ __forceinline__ bf16 f2bf(float f) { return (bf16)f; }

__device__ __forceinline__ v8f vzero8() {
  v8f z;
#pragma unroll
  for (int i = 0; i < 8; ++i) z[i] = 0.0f;
  return z;
}

__device__ __forceinline__ v8f wmma_bf16(v16bf a, v16bf b, v8f c) {
  // D = A(16x32 bf16) * B(32x16 bf16) + C(16x16 f32)
  return __builtin_amdgcn_wmma_f32_16x16x32_bf16(false, a, false, b,
                                                 (short)0, c, false, false);
}

// A-fragment (ISA 7.12.2, 16-bit A 16x32): lane L holds row M=(L&15);
// lanes 0-15: K={0..7,16..23}, lanes 16-31: K={8..15,24..31}.
__device__ __forceinline__ v16bf frag_a(const bf16* rowp, int lane) {
  const int kb = (lane & 16) ? 8 : 0;
  v8bf lo = *(const v8bf*)(rowp + kb);
  v8bf hi = *(const v8bf*)(rowp + 16 + kb);
  v16bf r;
#pragma unroll
  for (int i = 0; i < 8; ++i) { r[i] = lo[i]; r[i + 8] = hi[i]; }
  return r;
}

// B-fragment (16-bit B 32x16): lane L holds column N=(L&15);
// lanes 0-15: K={0..7,16..23}, lanes 16-31: K={8..15,24..31}.
__device__ __forceinline__ v16bf frag_b(const bf16* colp, int lane) {
  const int kb = (lane & 16) ? 8 : 0;
  v8bf lo = *(const v8bf*)(colp + kb);
  v8bf hi = *(const v8bf*)(colp + 16 + kb);
  v16bf r;
#pragma unroll
  for (int i = 0; i < 8; ++i) { r[i] = lo[i]; r[i + 8] = hi[i]; }
  return r;
}

// ---------------- Tensor Data Mover (async 2D tile global -> LDS) -----------
#if __has_builtin(__builtin_amdgcn_tensor_load_to_lds) && \
    __has_builtin(__builtin_amdgcn_s_wait_tensorcnt)
#define HAVE_TDM 1
typedef __attribute__((ext_vector_type(4))) unsigned int u32x4;
typedef __attribute__((ext_vector_type(8))) int i32x8;
typedef __attribute__((ext_vector_type(4))) int i32x4;

// D# per ISA cdna5 ch.8: group0 = {count/type, lds_addr, global_addr},
// group1 = {data_size/pad, tensor dims, tile dims, dim0 stride}.
// 2-byte elements; LDS row padding via pad_interval/pad_amount codes.
__device__ __forceinline__ void tdm_load_2d_bf16(
    unsigned lds_off, const void* gptr,
    int tile_x, int tile_y, int tensor_x, int tensor_y, int stride_x,
    int pad_interval_code, int pad_amount_code, int pad_enable)
{
  const unsigned long long ga = (unsigned long long)(uintptr_t)gptr;
  u32x4 g0;
  g0[0] = 1u;                                         // count=1, user D#
  g0[1] = lds_off;                                    // lds_addr (bytes)
  g0[2] = (unsigned)(ga & 0xffffffffu);               // global_addr[31:0]
  g0[3] = (unsigned)((ga >> 32) & 0x01ffffffu)        // global_addr[56:32]
        | (2u << 30);                                 // type=2 ("image")
  i32x8 g1;
  g1[0] = (1 << 16)                                   // data_size: 2 bytes
        | (pad_enable << 20)
        | (pad_interval_code << 22)
        | (pad_amount_code << 25);
  g1[1] = (tensor_x & 0xffff) << 16;                  // tensor_dim0[15:0]
  g1[2] = ((tensor_x >> 16) & 0xffff)                 // tensor_dim0[31:16]
        | ((tensor_y & 0xffff) << 16);                // tensor_dim1[15:0]
  g1[3] = ((tensor_y >> 16) & 0xffff)                 // tensor_dim1[31:16]
        | ((tile_x & 0xffff) << 16);                  // tile_dim0
  g1[4] = (tile_y & 0xffff);                          // tile_dim1 (tile_dim2=0)
  g1[5] = stride_x;                                   // tensor_dim0_stride lo32
  g1[6] = 0;
  g1[7] = 0;
  i32x4 z4 = {0, 0, 0, 0};
#if defined(__clang_major__) && (__clang_major__ >= 23)
  i32x8 z8 = {0, 0, 0, 0, 0, 0, 0, 0};
  __builtin_amdgcn_tensor_load_to_lds(g0, g1, z4, z4, z8, 0);
#else
  __builtin_amdgcn_tensor_load_to_lds(g0, g1, z4, z4, 0);
#endif
}
#else
#define HAVE_TDM 0
#warning "gfx1250 probe: tensor_load_to_lds / s_wait_tensorcnt NOT available; manual K staging"
#endif

#define EPI_F32           0   // Out.f32 = acc (+ Resid)
#define EPI_SILU_MUL_BF16 2   // Out.bf16 = silu(Aux) * acc

// ---- GEMM staging helpers (register pipeline) ------------------------------
__device__ __forceinline__ void stage_load(
    const bf16* __restrict__ A, const float* __restrict__ Bw,
    int m0, int n0, int K, int N, int k0, int tid,
    v8bf& areg, float4* breg)
{
  const int arow = tid >> 2, ac = tid & 3;
  areg = *(const v8bf*)(A + (size_t)(m0 + arow) * K + k0 + ac * 8);
#pragma unroll
  for (int it = 0; it < 4; ++it) {
    const int i = tid + it * 256;
    const int kk = i >> 5, c = i & 31;
    breg[it] = *(const float4*)(Bw + (size_t)(k0 + kk) * N + n0 + c * 4);
  }
}

__device__ __forceinline__ void stage_store(
    int tid, const v8bf& areg, const float4* breg,
    bf16 (*as_)[40], bf16 (*bsT)[48])
{
  const int arow = tid >> 2, ac = tid & 3;
  *(v8bf*)&as_[arow][ac * 8] = areg;
#pragma unroll
  for (int it = 0; it < 4; ++it) {
    const int i = tid + it * 256;
    const int kk = i >> 5, c = i & 31;
    const v2bf p01 = f2bf2(breg[it].x, breg[it].y);
    const v2bf p23 = f2bf2(breg[it].z, breg[it].w);
    bsT[c * 4 + 0][kk] = p01[0];
    bsT[c * 4 + 1][kk] = p01[1];
    bsT[c * 4 + 2][kk] = p23[0];
    bsT[c * 4 + 3][kk] = p23[1];
  }
}

// C[M,N] f32-accum = A[M,K] bf16 x Bw[K,N] f32(converted to bf16 on stage).
// Block tile 64x128, 8 waves (2x4), each wave 32x32 via 2x2 WMMA tiles.
// blockIdx.x = M (fast) so weight panels are L2-resident across M-blocks.
// Register pipeline: next tile's global loads issue before the WMMA block.
__global__ __launch_bounds__(256) void gemm_kernel(
    const bf16* __restrict__ A, const float* __restrict__ Bw,
    void* __restrict__ Out, const float* __restrict__ Resid,
    const float* __restrict__ Aux, int M, int N, int K, int epi)
{
  __shared__ __align__(32) bf16 as_[64][40];    // 64x32 A tile (+pad)
  __shared__ __align__(32) bf16 bsT[128][48];   // 32x128 B tile, K-contiguous

  const int tid  = threadIdx.x;
  const int lane = tid & 31;
  const int wave = tid >> 5;
  const int wm = wave >> 2;                     // 0..1
  const int wn = wave & 3;                      // 0..3
  const int m0 = blockIdx.x * 64;
  const int n0 = blockIdx.y * 128;
  const int cl  = lane & 15;
  const int rhi = (lane & 16) ? 8 : 0;

  v8f acc[2][2];
  acc[0][0] = vzero8(); acc[0][1] = vzero8();
  acc[1][0] = vzero8(); acc[1][1] = vzero8();

  v8bf areg;
  float4 breg[4];
  stage_load(A, Bw, m0, n0, K, N, 0, tid, areg, breg);

  for (int k0 = 0; k0 < K; k0 += 32) {
    stage_store(tid, areg, breg, as_, bsT);   // waits for in-flight loads here
    __syncthreads();
    if (k0 + 32 < K)                          // issue next tile's loads early
      stage_load(A, Bw, m0, n0, K, N, k0 + 32, tid, areg, breg);
    const v16bf af0 = frag_a(&as_[wm * 32 + cl][0], lane);
    const v16bf af1 = frag_a(&as_[wm * 32 + 16 + cl][0], lane);
    const v16bf bf0 = frag_b(&bsT[wn * 32 + cl][0], lane);
    const v16bf bf1 = frag_b(&bsT[wn * 32 + 16 + cl][0], lane);
    acc[0][0] = wmma_bf16(af0, bf0, acc[0][0]);
    acc[0][1] = wmma_bf16(af0, bf1, acc[0][1]);
    acc[1][0] = wmma_bf16(af1, bf0, acc[1][0]);
    acc[1][1] = wmma_bf16(af1, bf1, acc[1][1]);
    __syncthreads();                          // consumers done before re-store
  }

#pragma unroll
  for (int tm = 0; tm < 2; ++tm)
#pragma unroll
    for (int tn = 0; tn < 2; ++tn)
#pragma unroll
      for (int j = 0; j < 8; ++j) {
        const int row = m0 + wm * 32 + tm * 16 + j + rhi;
        const int col = n0 + wn * 32 + tn * 16 + cl;
        const size_t idx = (size_t)row * N + col;
        float v = acc[tm][tn][j];
        if (epi == EPI_F32) {
          if (Resid) v += Resid[idx];
          ((float*)Out)[idx] = v;
        } else { // EPI_SILU_MUL_BF16
          const float g  = Aux[idx];
          const float sg = g / (1.0f + __expf(-g));
          ((bf16*)Out)[idx] = f2bf(sg * v);
        }
      }
}

__global__ __launch_bounds__(256) void rmsnorm_kernel(
    const float* __restrict__ X, const float* __restrict__ W,
    bf16* __restrict__ Y)
{
  const int row = blockIdx.x;
  const float* x = X + (size_t)row * HID_;
  float ss = 0.f;
  for (int i = threadIdx.x; i < HID_ / 4; i += 256) {
    const float4 v = ((const float4*)x)[i];
    ss += v.x * v.x + v.y * v.y + v.z * v.z + v.w * v.w;
  }
#pragma unroll
  for (int m = 16; m >= 1; m >>= 1) ss += __shfl_xor(ss, m, 32);
  __shared__ float red[8];
  if ((threadIdx.x & 31) == 0) red[threadIdx.x >> 5] = ss;
  __syncthreads();
  float tot = 0.f;
#pragma unroll
  for (int i = 0; i < 8; ++i) tot += red[i];
  const float inv = rsqrtf(tot / (float)HID_ + 1e-6f);
  for (int i = threadIdx.x; i < HID_ / 4; i += 256) {
    const float4 v = ((const float4*)x)[i];
    const float4 w = ((const float4*)W)[i];
    v2bf* y = (v2bf*)(Y + (size_t)row * HID_ + i * 4);
    y[0] = f2bf2(v.x * inv * w.x, v.y * inv * w.y);
    y[1] = f2bf2(v.z * inv * w.z, v.w * inv * w.w);
  }
}

// RoPE on q/k, convert all of q/k/v to bf16 in [B,H,S,HD] layout.
__global__ __launch_bounds__(256) void rope_pack_kernel(
    const float* __restrict__ Qf, const float* __restrict__ Kf,
    const float* __restrict__ Vf, const int* __restrict__ Pos,
    bf16* __restrict__ Qbf, bf16* __restrict__ Kbf, bf16* __restrict__ Vbf)
{
  const int tok = blockIdx.x;
  const int b = tok >> 10, s = tok & (S_ - 1);
  const float pos = (float)Pos[tok];
  const float* q = Qf + (size_t)tok * (NH_ * HD_);
  const float* k = Kf + (size_t)tok * (NKV_ * HD_);
  const float* v = Vf + (size_t)tok * (NKV_ * HD_);
  const float lnth = 9.210340371976184f / 64.0f;   // ln(10000)/64
  for (int i = threadIdx.x; i < NH_ * HD_; i += 256) {
    const int h = i >> 7, d = i & 127, j = d & 63;
    const float ang = pos * __expf(-(float)j * lnth);
    const float c = __cosf(ang), sn = __sinf(ang);
    const float rot = (d < 64) ? -q[i + 64] : q[i - 64];
    Qbf[(((size_t)b * NH_ + h) * S_ + s) * HD_ + d] = f2bf(q[i] * c + rot * sn);
  }
  for (int i = threadIdx.x; i < NKV_ * HD_; i += 256) {
    const int h = i >> 7, d = i & 127, j = d & 63;
    const float ang = pos * __expf(-(float)j * lnth);
    const float c = __cosf(ang), sn = __sinf(ang);
    const float rot = (d < 64) ? -k[i + 64] : k[i - 64];
    Kbf[(((size_t)b * NKV_ + h) * S_ + s) * HD_ + d] = f2bf(k[i] * c + rot * sn);
    Vbf[(((size_t)b * NKV_ + h) * S_ + s) * HD_ + d] = f2bf(v[i]);
  }
}

// Flash attention: 128 q-rows per block (8 waves x 16 rows), causal, GQA 4:1.
// K tile via TDM (async DMA, LDS row padding in-descriptor), V^T staged
// manually in parallel, bf16 WMMA for both GEMMs, fp32 online softmax.
__global__ __launch_bounds__(256) void attn_kernel(
    const bf16* __restrict__ Qbf, const bf16* __restrict__ Kbf,
    const bf16* __restrict__ Vbf, bf16* __restrict__ Obf)
{
  const int qblk = blockIdx.x;            // S/128
  const int h    = blockIdx.y;            // NH
  const int b    = blockIdx.z;            // B
  const int hk   = h >> 2;                // NH/NKV = 4

  const int tid = threadIdx.x, lane = tid & 31, wave = tid >> 5;
  const int cl  = lane & 15;
  const int rhi = (lane & 16) ? 8 : 0;

  __shared__ __align__(32) bf16 ks [64][HD_ + 16];   // K tile (row 256B+32B pad)
  __shared__ __align__(32) bf16 vsT[HD_][64 + 16];   // V tile, transposed
  __shared__ __align__(32) bf16 pb [8][16][64 + 8];  // per-wave P transpose buf

  const bf16* qrow =
      Qbf + (((size_t)b * NH_ + h) * S_ + qblk * 128 + wave * 16 + cl) * HD_;
  v16bf qf[4];
#pragma unroll
  for (int c = 0; c < 4; ++c) qf[c] = frag_a(qrow + c * 32, lane);

  float mrun[8], lrun[8];
#pragma unroll
  for (int j = 0; j < 8; ++j) { mrun[j] = -1e30f; lrun[j] = 0.f; }
  v8f oacc[8];
#pragma unroll
  for (int t = 0; t < 8; ++t) oacc[t] = vzero8();

  const bf16* kbase = Kbf + (((size_t)b * NKV_ + hk) * S_) * HD_;
  const bf16* vbase = Vbf + (((size_t)b * NKV_ + hk) * S_) * HD_;
  const float scale = 0.08838834764831845f;   // 1/sqrt(128)
  const int ktiles = 2 * qblk + 2;            // causal: keys <= qblk*128+127

  for (int kt = 0; kt < ktiles; ++kt) {
    __syncthreads();   // all waves done reading previous ks/vsT
#if HAVE_TDM
    if (wave == 0) {
      // 64x128 bf16 tile; row 256B = 64 DWORDs -> pad_interval code 5,
      // pad 8 DWORDs (32B = 16 bf16) -> pad_amount code 7 => LDS pitch 144.
      tdm_load_2d_bf16((unsigned)(uintptr_t)&ks[0][0],
                       kbase + (size_t)(kt * 64) * HD_,
                       /*tile_x=*/HD_, /*tile_y=*/64,
                       /*tensor_x=*/HD_, /*tensor_y=*/S_,
                       /*stride_x=*/HD_, 5, 7, 1);
    }
#else
    for (int i = tid; i < 64 * 16; i += 256) {
      const int row = i >> 4, c = i & 15;
      *(v8bf*)&ks[row][c * 8] =
          *(const v8bf*)(kbase + (size_t)(kt * 64 + row) * HD_ + c * 8);
    }
#endif
    for (int i = tid; i < 64 * 16; i += 256) {  // V^T staging (all waves)
      const int key = i >> 4, c = i & 15;
      const v8bf x = *(const v8bf*)(vbase + (size_t)(kt * 64 + key) * HD_ + c * 8);
#pragma unroll
      for (int jj = 0; jj < 8; ++jj) vsT[c * 8 + jj][key] = x[jj];
    }
#if HAVE_TDM
    if (wave == 0) __builtin_amdgcn_s_wait_tensorcnt(0);
#endif
    __syncthreads();

    // scores: 16(q rows of this wave) x 64(keys)
    v8f sc[4];
#pragma unroll
    for (int tn = 0; tn < 4; ++tn) {
      sc[tn] = vzero8();
#pragma unroll
      for (int c = 0; c < 4; ++c)
        sc[tn] = wmma_bf16(qf[c], frag_b(&ks[tn * 16 + cl][c * 32], lane), sc[tn]);
    }
    // scale + causal mask + row max
    float mt[8];
#pragma unroll
    for (int j = 0; j < 8; ++j) mt[j] = -1e30f;
#pragma unroll
    for (int tn = 0; tn < 4; ++tn) {
      const int kcol = kt * 64 + tn * 16 + cl;
#pragma unroll
      for (int j = 0; j < 8; ++j) {
        const int qrow_g = qblk * 128 + wave * 16 + j + rhi;
        float vv = sc[tn][j] * scale;
        vv = (kcol <= qrow_g) ? vv : -1e30f;
        sc[tn][j] = vv;
        mt[j] = fmaxf(mt[j], vv);
      }
    }
#pragma unroll
    for (int j = 0; j < 8; ++j) {
#pragma unroll
      for (int m = 8; m >= 1; m >>= 1)   // reduce within 16-lane half
        mt[j] = fmaxf(mt[j], __shfl_xor(mt[j], m, 32));
      const float mnew = fmaxf(mrun[j], mt[j]);
      const float corr = __expf(mrun[j] - mnew);
      mrun[j] = mnew;
      lrun[j] *= corr;
#pragma unroll
      for (int t = 0; t < 8; ++t) oacc[t][j] *= corr;
    }
    float rs[8];
#pragma unroll
    for (int j = 0; j < 8; ++j) rs[j] = 0.f;
#pragma unroll
    for (int tn = 0; tn < 4; ++tn)
#pragma unroll
      for (int j = 0; j < 8; ++j) {
        const float p = __expf(sc[tn][j] - mrun[j]);
        rs[j] += p;
        pb[wave][j + rhi][tn * 16 + cl] = f2bf(p);  // transpose via LDS
      }
#pragma unroll
    for (int j = 0; j < 8; ++j) {
#pragma unroll
      for (int m = 8; m >= 1; m >>= 1) rs[j] += __shfl_xor(rs[j], m, 32);
      lrun[j] += rs[j];
    }
    // P(16x64) x V(64x128); DS ops from same wave are in-order
#pragma unroll
    for (int c = 0; c < 2; ++c) {
      const v16bf pf = frag_a(&pb[wave][cl][c * 32], lane);
#pragma unroll
      for (int t = 0; t < 8; ++t)
        oacc[t] = wmma_bf16(pf, frag_b(&vsT[t * 16 + cl][c * 32], lane), oacc[t]);
    }
  }

  // normalize and store token-major [B*S, NH*HD] bf16
#pragma unroll
  for (int t = 0; t < 8; ++t)
#pragma unroll
    for (int j = 0; j < 8; ++j) {
      const int row = qblk * 128 + wave * 16 + j + rhi;
      const size_t tokidx = (size_t)b * S_ + row;
      Obf[tokidx * (NH_ * HD_) + h * HD_ + t * 16 + cl] =
          f2bf(oacc[t][j] / lrun[j]);
    }
}

extern "C" void kernel_launch(void* const* d_in, const int* in_sizes, int n_in,
                              void* d_out, int out_size, void* d_ws, size_t ws_size,
                              hipStream_t stream)
{
  (void)in_sizes; (void)n_in; (void)out_size; (void)ws_size;
  const float* hidden = (const float*)d_in[0];
  const int*   pos    = (const int*)d_in[1];
  // d_in[2] attn_mask: causal mask computed analytically in attn_kernel
  const float* Wq  = (const float*)d_in[3];
  const float* Wk  = (const float*)d_in[4];
  const float* Wv  = (const float*)d_in[5];
  const float* Wo  = (const float*)d_in[6];
  const float* ln1 = (const float*)d_in[7];
  const float* ln2 = (const float*)d_in[8];
  const float* Wg  = (const float*)d_in[9];
  const float* Wu  = (const float*)d_in[10];
  const float* Wd  = (const float*)d_in[11];
  float* out = (float*)d_out;

  char* ws = (char*)d_ws;
  const size_t TH  = (size_t)TOK_ * HID_;         // 16.78M elems
  const size_t TKV = (size_t)TOK_ * NKV_ * HD_;   // 4.19M elems
  const size_t TF  = (size_t)TOK_ * FF_;          // 45.1M elems
  size_t o = 0;
  bf16*  x1     = (bf16*)(ws + o);  o += TH * 2;
  const size_t o_alias = o;                       // start of attention scratch
  float* qf32   = (float*)(ws + o); o += TH * 4;
  float* kf32   = (float*)(ws + o); o += TKV * 4;
  float* vf32   = (float*)(ws + o); o += TKV * 4;
  bf16*  qbf    = (bf16*)(ws + o);  o += TH * 2;
  bf16*  kbf    = (bf16*)(ws + o);  o += TKV * 2;
  bf16*  vbf    = (bf16*)(ws + o);  o += TKV * 2;
  bf16*  attnbf = (bf16*)(ws + o);  o += TH * 2;
  float* hf32   = (float*)(ws + o); o += TH * 4;
  bf16*  x2     = (bf16*)(ws + o);  o += TH * 2;
  bf16*  gubf   = (bf16*)(ws + o);  o += TF * 2;
  // gate f32 (180MB) aliases q/k/v/attn scratch (184MB, dead by MLP phase)
  float* gf32   = (float*)(ws + o_alias);

  rmsnorm_kernel<<<TOK_, 256, 0, stream>>>(hidden, ln1, x1);
  gemm_kernel<<<dim3(TOK_/64, (NH_*HD_)/128), 256, 0, stream>>>(
      x1, Wq, qf32, nullptr, nullptr, TOK_, NH_*HD_, HID_, EPI_F32);
  gemm_kernel<<<dim3(TOK_/64, (NKV_*HD_)/128), 256, 0, stream>>>(
      x1, Wk, kf32, nullptr, nullptr, TOK_, NKV_*HD_, HID_, EPI_F32);
  gemm_kernel<<<dim3(TOK_/64, (NKV_*HD_)/128), 256, 0, stream>>>(
      x1, Wv, vf32, nullptr, nullptr, TOK_, NKV_*HD_, HID_, EPI_F32);
  rope_pack_kernel<<<TOK_, 256, 0, stream>>>(qf32, kf32, vf32, pos, qbf, kbf, vbf);
  attn_kernel<<<dim3(S_/128, NH_, B_), 256, 0, stream>>>(qbf, kbf, vbf, attnbf);
  gemm_kernel<<<dim3(TOK_/64, HID_/128), 256, 0, stream>>>(
      attnbf, Wo, hf32, hidden, nullptr, TOK_, HID_, NH_*HD_, EPI_F32);
  rmsnorm_kernel<<<TOK_, 256, 0, stream>>>(hf32, ln2, x2);
  gemm_kernel<<<dim3(TOK_/64, FF_/128), 256, 0, stream>>>(
      x2, Wg, gf32, nullptr, nullptr, TOK_, FF_, HID_, EPI_F32);
  gemm_kernel<<<dim3(TOK_/64, FF_/128), 256, 0, stream>>>(
      x2, Wu, gubf, nullptr, gf32, TOK_, FF_, HID_, EPI_SILU_MUL_BF16);
  gemm_kernel<<<dim3(TOK_/64, HID_/128), 256, 0, stream>>>(
      gubf, Wd, out, hf32, nullptr, TOK_, HID_, FF_, EPI_F32);
}
